// PixelContrastLoss_10376640987313
// MI455X (gfx1250) — matrix operand order repaired and checked by reading, not compiled, not run on hardware
//
#include <hip/hip_runtime.h>

typedef __attribute__((ext_vector_type(16))) __bf16   v16bf;
typedef __attribute__((ext_vector_type(8)))  float    v8f;
typedef __attribute__((ext_vector_type(4)))  unsigned u32x4;
typedef __attribute__((ext_vector_type(4)))  float    f32x4;
typedef __attribute__((ext_vector_type(2)))  unsigned u32x2;

#define TEMP_     0.1f
#define BASETEMP_ 0.07f
#define B_   8
#define NP_  4096
#define NS_  2048
#define D_   256

// ---------------- fp32 -> bf16 (RNE) conversion ----------------
__device__ __forceinline__ unsigned f2bf1(float f) {
    unsigned u = __float_as_uint(f);
    return (u + 0x7FFFu + ((u >> 16) & 1u)) >> 16;
}

__global__ void pcl_zero_out(float* out) { out[0] = 0.0f; }

__global__ __launch_bounds__(256) void pcl_cvt_bf16x4(const float* __restrict__ src,
                                                      unsigned* __restrict__ dst, int n4) {
    int i = blockIdx.x * blockDim.x + threadIdx.x;
    if (i >= n4) return;
    f32x4 f = ((const f32x4*)src)[i];
    u32x2 o;
    o.x = f2bf1(f.x) | (f2bf1(f.y) << 16);
    o.y = f2bf1(f.z) | (f2bf1(f.w) << 16);
    ((u32x2*)dst)[i] = o;
}

// ---------------- cross-lane reductions (within 16-lane groups of a wave32) ----
__device__ __forceinline__ float rmax16(float x) {
    x = fmaxf(x, __shfl_xor(x, 1, 32));
    x = fmaxf(x, __shfl_xor(x, 2, 32));
    x = fmaxf(x, __shfl_xor(x, 4, 32));
    x = fmaxf(x, __shfl_xor(x, 8, 32));
    return x;
}
__device__ __forceinline__ float rsum16(float x) {
    x += __shfl_xor(x, 1, 32);
    x += __shfl_xor(x, 2, 32);
    x += __shfl_xor(x, 4, 32);
    x += __shfl_xor(x, 8, 32);
    return x;
}

// ---------------- fused GEMM + masked softmax-contrast reduction ----------------
// One block = 8 wave32s handles (batch b, 16 pixel rows). Each wave owns 256 seg
// columns as 16 WMMA accumulator tiles (v8f each) that stay register-resident.
__global__ __launch_bounds__(256) void pcl_main(
    const unsigned short* __restrict__ Pbf,   // [B][NP][D] bf16
    const unsigned short* __restrict__ Sbf,   // [B][NS][D] bf16
    const int* __restrict__ pix_label,        // [B][NP]
    const int* __restrict__ seg_label,        // [B][NS]
    float* __restrict__ out)
{
    __shared__ float lmax[8][16];
    __shared__ float lneg[8][16];
    __shared__ float lpos[8][16];
    __shared__ float lcnt[8][16];

    const int tid  = threadIdx.x;
    const int wave = tid >> 5;
    const int lane = tid & 31;
    const int l15  = lane & 15;
    const int hi   = lane >> 4;

    const int b  = blockIdx.x / (NP_ / 16);
    const int p0 = (blockIdx.x % (NP_ / 16)) * 16;
    const int s0 = wave * 256;                 // this wave's segment-column base

    v8f acc[16] = {};

    // A fragment (16x32 bf16): lane l -> row (l&15); lanes<16 hold K[0..16),
    // lanes>=16 hold K[16..32) -> one contiguous 32B (two b128 loads).
    const unsigned short* aBase = Pbf + (size_t)(b * NP_ + p0 + l15) * D_ + hi * 16;
    // B fragment (32x16 bf16): lane l -> col (l&15); lanes<16 hold K[0..16),
    // lanes>=16 hold K[16..32) -> one contiguous 32B (two b128 loads).
    const unsigned short* bBase = Sbf + (size_t)(b * NS_ + s0 + l15) * D_ + hi * 16;

    for (int ks = 0; ks < D_ / 32; ++ks) {      // K loop: 256 / 32 = 8 steps
        union { u32x4 q[2]; v16bf v; } A;
        const u32x4* ap = (const u32x4*)(aBase + ks * 32);
        A.q[0] = ap[0];
        A.q[1] = ap[1];                          // +8 elements (K+8)
        #pragma unroll
        for (int nt = 0; nt < 16; ++nt) {
            union { u32x4 q[2]; v16bf v; } Bf;
            const u32x4* bp = (const u32x4*)(bBase + (size_t)nt * 16 * D_ + ks * 32);
            Bf.q[0] = bp[0];
            Bf.q[1] = bp[1];
            acc[nt] = __builtin_amdgcn_wmma_f32_16x16x32_bf16(
                false, A.v, false, Bf.v, (short)0, acc[nt], false, false);
        }
    }

    // Labels. C/D layout: lane holds col n = s0 + nt*16 + (l&15); VGPR v holds row v + 8*hi.
    int plab[8];
    #pragma unroll
    for (int v = 0; v < 8; ++v) plab[v] = pix_label[b * NP_ + p0 + hi * 8 + v];
    int slab[16];
    #pragma unroll
    for (int nt = 0; nt < 16; ++nt) slab[nt] = seg_label[b * NS_ + s0 + nt * 16 + l15];

    const float invT = 1.0f / TEMP_;

    // ---- pass 1: per-row max over all 2048 columns ----
    float rowm[8];
    #pragma unroll
    for (int v = 0; v < 8; ++v) rowm[v] = -3.4e38f;
    #pragma unroll
    for (int nt = 0; nt < 16; ++nt)
        #pragma unroll
        for (int v = 0; v < 8; ++v) rowm[v] = fmaxf(rowm[v], acc[nt][v] * invT);
    #pragma unroll
    for (int v = 0; v < 8; ++v) rowm[v] = rmax16(rowm[v]);
    if (l15 == 0) {
        #pragma unroll
        for (int v = 0; v < 8; ++v) lmax[wave][hi * 8 + v] = rowm[v];
    }
    __syncthreads();
    float mx[8];
    #pragma unroll
    for (int v = 0; v < 8; ++v) {
        float m = lmax[0][hi * 8 + v];
        #pragma unroll
        for (int w = 1; w < 8; ++w) m = fmaxf(m, lmax[w][hi * 8 + v]);
        mx[v] = m;
    }

    // ---- pass 2: neg = sum over label-mismatched columns of exp(logit - max) ----
    float nsum[8] = {};
    #pragma unroll
    for (int nt = 0; nt < 16; ++nt) {
        int sl = slab[nt];
        #pragma unroll
        for (int v = 0; v < 8; ++v) {
            float lg = acc[nt][v] * invT - mx[v];
            float e  = __expf(lg);
            nsum[v] += (sl != plab[v]) ? e : 0.0f;
        }
    }
    #pragma unroll
    for (int v = 0; v < 8; ++v) nsum[v] = rsum16(nsum[v]);
    if (l15 == 0) {
        #pragma unroll
        for (int v = 0; v < 8; ++v) lneg[wave][hi * 8 + v] = nsum[v];
    }
    __syncthreads();
    float ng[8];
    #pragma unroll
    for (int v = 0; v < 8; ++v) {
        float s = 0.0f;
        #pragma unroll
        for (int w = 0; w < 8; ++w) s += lneg[w][hi * 8 + v];
        ng[v] = s;
    }

    // ---- pass 3: sum over positives of (lg - log(exp(lg) + neg)), and counts ----
    float psum[8] = {}, pcnt[8] = {};
    #pragma unroll
    for (int nt = 0; nt < 16; ++nt) {
        int sl = slab[nt];
        #pragma unroll
        for (int v = 0; v < 8; ++v) {
            float lg  = acc[nt][v] * invT - mx[v];
            bool  pos = (sl == plab[v]);
            float lp  = lg - __logf(__expf(lg) + ng[v]);
            psum[v] += pos ? lp : 0.0f;
            pcnt[v] += pos ? 1.0f : 0.0f;
        }
    }
    #pragma unroll
    for (int v = 0; v < 8; ++v) { psum[v] = rsum16(psum[v]); pcnt[v] = rsum16(pcnt[v]); }
    if (l15 == 0) {
        #pragma unroll
        for (int v = 0; v < 8; ++v) { lpos[wave][hi * 8 + v] = psum[v]; lcnt[wave][hi * 8 + v] = pcnt[v]; }
    }
    __syncthreads();

    // ---- finalize 16 rows, accumulate into global scalar ----
    if (tid < 16) {
        float P = 0.0f, C = 0.0f;
        #pragma unroll
        for (int w = 0; w < 8; ++w) { P += lpos[w][tid]; C += lcnt[w][tid]; }
        float pa = -(TEMP_ / BASETEMP_) * (P / C);
        pa = rsum16(pa);
        if (tid == 0) atomicAdd(out, pa * (1.0f / (B_ * (float)NP_)));
    }
}

// ---------------- launcher ----------------
extern "C" void kernel_launch(void* const* d_in, const int* in_sizes, int n_in,
                              void* d_out, int out_size, void* d_ws, size_t ws_size,
                              hipStream_t stream) {
    const float* seg_feature = (const float*)d_in[0];
    const int*   seg_label   = (const int*)d_in[1];
    const float* pix_feature = (const float*)d_in[2];
    const int*   pix_label   = (const int*)d_in[3];
    float* out = (float*)d_out;

    // workspace: bf16 pix features (16 MB) then bf16 seg features (8 MB)
    unsigned short* Pbf = (unsigned short*)d_ws;
    unsigned short* Sbf = (unsigned short*)((char*)d_ws + (size_t)B_ * NP_ * D_ * sizeof(unsigned short));

    pcl_zero_out<<<1, 1, 0, stream>>>(out);

    int np4 = B_ * NP_ * D_ / 4;
    int ns4 = B_ * NS_ * D_ / 4;
    pcl_cvt_bf16x4<<<(np4 + 255) / 256, 256, 0, stream>>>(pix_feature, (unsigned*)Pbf, np4);
    pcl_cvt_bf16x4<<<(ns4 + 255) / 256, 256, 0, stream>>>(seg_feature, (unsigned*)Sbf, ns4);

    pcl_main<<<B_ * (NP_ / 16), 256, 0, stream>>>(Pbf, Sbf, pix_label, seg_label, out);
}
